// Attacker_1606317769452
// MI455X (gfx1250) — matrix operand order repaired and checked by reading, not compiled, not run on hardware
//
#include <hip/hip_runtime.h>
#include <hip/hip_bf16.h>

// ---------------- problem constants ----------------
constexpr int N_NODES = 100000;
constexpr int N_EDGES = 1600000;
constexpr int F_IN  = 512;
constexpr int H1R   = 180, H1P = 192;   // real / padded
constexpr int H2R   = 120, H2P = 128;
constexpr int NCR   = 40,  NCP = 48;
constexpr float NEG_SLOPE = 0.01f;

constexpr int MT = N_NODES / 16;        // 6250 row tiles (exact)

typedef __attribute__((ext_vector_type(16))) __bf16 v16bf;
typedef __attribute__((ext_vector_type(8)))  float  v8f;

#if defined(__has_builtin)
#if __has_builtin(__builtin_amdgcn_sched_barrier)
#define SCHED_FENCE() __builtin_amdgcn_sched_barrier(0)
#endif
#endif
#ifndef SCHED_FENCE
#define SCHED_FENCE()
#endif

// round-to-nearest-even fp32 -> bf16 bits
__device__ __forceinline__ unsigned short f2bf(float f) {
  union { float f; unsigned u; } x; x.f = f;
  unsigned u = x.u;
  unsigned r = u + 0x7fffu + ((u >> 16) & 1u);
  return (unsigned short)(r >> 16);
}

// ---------------- degree / norm ----------------
__global__ void k_fill1(float* __restrict__ p, int n) {
  int i = blockIdx.x * blockDim.x + threadIdx.x;
  if (i < n) p[i] = 1.0f;                       // self loop contributes 1 to deg
}
__global__ void k_deg_scatter(const int* __restrict__ dst, float* __restrict__ deg) {
  int e = blockIdx.x * blockDim.x + threadIdx.x;
  if (e < N_EDGES) atomicAdd(&deg[dst[e]], 1.0f);
}
__global__ void k_dinv(float* __restrict__ d, int n) {
  int i = blockIdx.x * blockDim.x + threadIdx.x;
  if (i < n) { float v = d[i]; d[i] = v > 0.f ? rsqrtf(v) : 0.f; }
}

__global__ void k_zero(float* __restrict__ p, long long n) {
  long long i = (long long)blockIdx.x * blockDim.x + threadIdx.x;
  if (i < n) p[i] = 0.f;
}

// ---------------- weight packing: B-fragment (32x16 bf16 per fragment) ----
// gemm reads: wfrag[((kb*NT+nt)*32 + lane)*16 .. +15]
// element 2v,2v+1 of lane l  <->  W[k0 + 2v + 16*(l>>4) + {0,1}][nt*16 + (l&15)]
__global__ void k_pack_w(const float* __restrict__ W, unsigned short* __restrict__ out,
                         int Kreal, int Nreal, int KT, int NT) {
  int idx = blockIdx.x * blockDim.x + threadIdx.x;
  if (idx >= KT * NT * 32) return;
  int lane = idx & 31;
  int frag = idx >> 5;
  int nt = frag % NT, kb = frag / NT;
  int half = lane >> 4, n = nt * 16 + (lane & 15);
  unsigned short* o = out + (size_t)idx * 16;
#pragma unroll
  for (int e = 0; e < 16; ++e) {
    int v = e >> 1, lo = e & 1;
    int k = kb * 32 + 2 * v + 16 * half + lo;
    float f = (k < Kreal && n < Nreal) ? W[(size_t)k * Nreal + n] : 0.f;
    o[e] = f2bf(f);
  }
}

// ---------------- activation packing: A-fragment (16x32 bf16) -------------
// gemm reads: afrag[((mt*KT+kb)*32 + lane)*16 .. +15]
// element 2j,2j+1 of lane l <-> A[mt*16 + (l&15)][k0 + kbase(j, l>>4) + {0,1}]
__global__ void k_pack_a(const float* __restrict__ A, unsigned short* __restrict__ out,
                         int KT) {
  long long idx = (long long)blockIdx.x * blockDim.x + threadIdx.x;
  long long total = (long long)MT * KT * 32;
  if (idx >= total) return;
  int lane = (int)(idx & 31);
  long long frag = idx >> 5;
  int kb = (int)(frag % KT);
  int mt = (int)(frag / KT);
  int half = lane >> 4;
  int K = KT * 32;
  const float* row = A + (size_t)(mt * 16 + (lane & 15)) * K + kb * 32;
  unsigned short* o = out + idx * 16;
#pragma unroll
  for (int e = 0; e < 16; ++e) {
    int j = e >> 1, lo = e & 1;
    int kk = 2 * j + 16 * half;
    o[e] = f2bf(row[kk + lo]);
  }
}

// ---------------- WMMA GEMM: C[M x NT*16] = Afrag @ Wfrag -----------------
// Per k-step: batch-load A + all NT B fragments (one long clause, ~2*(NT+1)
// b128s in flight), sched_barrier to stop the scheduler from sinking loads,
// then NT back-to-back v_wmma into the XDL pipe with a single wait.
template <int NT, int KT>
__global__ void __launch_bounds__(256)
k_gemm(const unsigned short* __restrict__ afrag,
       const unsigned short* __restrict__ wfrag,
       float* __restrict__ C) {
  int lane = threadIdx.x & 31;
  int mt = blockIdx.x * (blockDim.x >> 5) + (threadIdx.x >> 5);
  if (mt >= MT) return;                        // wave-uniform: EXEC all-1s for WMMA
  v8f acc[NT] = {};
  const v16bf* ap = (const v16bf*)(afrag) + ((size_t)mt * KT * 32 + lane);
  const v16bf* bp = (const v16bf*)(wfrag) + lane;
#pragma unroll 1
  for (int kb = 0; kb < KT; ++kb) {
    v16bf a = ap[(size_t)kb * 32];
    v16bf b[NT];
#pragma unroll
    for (int nt = 0; nt < NT; ++nt) b[nt] = bp[(size_t)(kb * NT + nt) * 32];
    SCHED_FENCE();                             // keep all loads batched before WMMAs
#pragma unroll
    for (int nt = 0; nt < NT; ++nt)
      acc[nt] = __builtin_amdgcn_wmma_f32_16x16x32_bf16(
          false, a, false, b[nt], (short)0, acc[nt], false, false);
    SCHED_FENCE();
  }
  const int Npad = NT * 16;
  int half = lane >> 4, n0 = lane & 15;
#pragma unroll
  for (int nt = 0; nt < NT; ++nt)
#pragma unroll
    for (int r = 0; r < 8; ++r)
      C[(size_t)(mt * 16 + r + 8 * half) * Npad + nt * 16 + n0] = acc[nt][r];
}

// ---------------- edge scatter-add (wave per edge) ------------------------
template <int F>
__global__ void k_scatter(const float* __restrict__ h, float* __restrict__ agg,
                          const int* __restrict__ src, const int* __restrict__ dst,
                          const float* __restrict__ dinv) {
  int lane = threadIdx.x & 31;
  long long e = (long long)blockIdx.x * (blockDim.x >> 5) + (threadIdx.x >> 5);
  if (e >= N_EDGES) return;
  int s = src[e], d = dst[e];
  const float* hs = h + (size_t)s * F;
  __builtin_prefetch(hs + lane * 4, 0, 1);     // global_prefetch: pull source row early
  float w = dinv[s] * dinv[d];
  float* ad = agg + (size_t)d * F;
#pragma unroll
  for (int f = lane; f < F; f += 32) atomicAdd(&ad[f], w * hs[f]);
}

// ---------------- fused self-loop + bias + leaky-relu (in place) ----------
template <int FP, int FR>
__global__ void k_post(float* __restrict__ agg, const float* __restrict__ h,
                       const float* __restrict__ dinv, const float* __restrict__ b) {
  long long idx = (long long)blockIdx.x * blockDim.x + threadIdx.x;
  if (idx >= (long long)N_NODES * FP) return;
  int row = (int)(idx / FP), col = (int)(idx % FP);
  float di = dinv[row];
  float v = agg[idx] + di * di * h[idx] + (col < FR ? b[col] : 0.f);
  agg[idx] = v >= 0.f ? v : NEG_SLOPE * v;
}

// ---------------- fused bias + log-softmax (wave per row) -----------------
__global__ void k_logsoftmax(const float* __restrict__ logits,
                             const float* __restrict__ b3,
                             float* __restrict__ out) {
  int lane = threadIdx.x & 31;
  int row = blockIdx.x * (blockDim.x >> 5) + (threadIdx.x >> 5);
  if (row >= N_NODES) return;
  const float* L = logits + (size_t)row * NCP;
  const float NEG = -3.4e38f;
  float v0 = (lane < NCR)      ? L[lane]      + b3[lane]      : NEG;
  float v1 = (lane + 32 < NCR) ? L[lane + 32] + b3[lane + 32] : NEG;
  float m = fmaxf(v0, v1);
#pragma unroll
  for (int off = 16; off; off >>= 1) m = fmaxf(m, __shfl_xor(m, off, 32));
  float s = ((lane < NCR) ? expf(v0 - m) : 0.f) +
            ((lane + 32 < NCR) ? expf(v1 - m) : 0.f);
#pragma unroll
  for (int off = 16; off; off >>= 1) s += __shfl_xor(s, off, 32);
  float ls = logf(s);
  if (lane < NCR)      out[(size_t)row * NCR + lane]      = v0 - m - ls;
  if (lane + 32 < NCR) out[(size_t)row * NCR + lane + 32] = v1 - m - ls;
}

// ---------------- launch ----------------
static inline size_t align_up(size_t x) { return (x + 4095) & ~size_t(4095); }
static inline int cdiv(long long a, long long b) { return (int)((a + b - 1) / b); }

extern "C" void kernel_launch(void* const* d_in, const int* in_sizes, int n_in,
                              void* d_out, int out_size, void* d_ws, size_t ws_size,
                              hipStream_t stream) {
  const float* x  = (const float*)d_in[0];
  const int*   ei = (const int*)d_in[1];
  const int*   src = ei;
  const int*   dst = ei + N_EDGES;
  const float* W1 = (const float*)d_in[2];
  const float* b1 = (const float*)d_in[3];
  const float* W2 = (const float*)d_in[4];
  const float* b2 = (const float*)d_in[5];
  const float* W3 = (const float*)d_in[6];
  const float* b3 = (const float*)d_in[7];
  float* out = (float*)d_out;

  // workspace carve-up
  char* ws = (char*)d_ws;
  size_t off = 0;
  float* dinv = (float*)(ws + off); off += align_up((size_t)N_NODES * 4);
  unsigned short* wf1 = (unsigned short*)(ws + off); off += align_up((size_t)16 * 12 * 32 * 16 * 2);
  unsigned short* wf2 = (unsigned short*)(ws + off); off += align_up((size_t)6  * 8  * 32 * 16 * 2);
  unsigned short* wf3 = (unsigned short*)(ws + off); off += align_up((size_t)4  * 3  * 32 * 16 * 2);
  unsigned short* bufA = (unsigned short*)(ws + off); off += align_up((size_t)N_NODES * F_IN * 2);
  float* bufH = (float*)(ws + off); off += align_up((size_t)N_NODES * H1P * 4);
  float* bufG = (float*)(ws + off); off += align_up((size_t)N_NODES * H1P * 4);
  (void)ws_size; (void)in_sizes; (void)n_in; (void)out_size;

  const int B = 256;

  // 1) degree -> dinv (self-loop folded in via init=1)
  k_fill1<<<cdiv(N_NODES, B), B, 0, stream>>>(dinv, N_NODES);
  k_deg_scatter<<<cdiv(N_EDGES, B), B, 0, stream>>>(dst, dinv);
  k_dinv<<<cdiv(N_NODES, B), B, 0, stream>>>(dinv, N_NODES);

  // 2) pack weights to bf16 B-fragments
  k_pack_w<<<cdiv(16 * 12 * 32, B), B, 0, stream>>>(W1, wf1, F_IN, H1R, 16, 12);
  k_pack_w<<<cdiv(6  * 8  * 32, B), B, 0, stream>>>(W2, wf2, H1P, H2R, 6, 8);
  k_pack_w<<<cdiv(4  * 3  * 32, B), B, 0, stream>>>(W3, wf3, H2P, NCR, 4, 3);

  // ---- layer 1 ----
  k_pack_a<<<cdiv((long long)MT * 16 * 32, B), B, 0, stream>>>(x, bufA, 16);
  k_gemm<12, 16><<<cdiv(MT, 8), B, 0, stream>>>(bufA, wf1, bufH);
  k_zero<<<cdiv((long long)N_NODES * H1P, B), B, 0, stream>>>(bufG, (long long)N_NODES * H1P);
  k_scatter<H1P><<<cdiv(N_EDGES, 8), B, 0, stream>>>(bufH, bufG, src, dst, dinv);
  k_post<H1P, H1R><<<cdiv((long long)N_NODES * H1P, B), B, 0, stream>>>(bufG, bufH, dinv, b1);

  // ---- layer 2 ----
  k_pack_a<<<cdiv((long long)MT * 6 * 32, B), B, 0, stream>>>(bufG, bufA, 6);
  k_gemm<8, 6><<<cdiv(MT, 8), B, 0, stream>>>(bufA, wf2, bufH);
  k_zero<<<cdiv((long long)N_NODES * H2P, B), B, 0, stream>>>(bufG, (long long)N_NODES * H2P);
  k_scatter<H2P><<<cdiv(N_EDGES, 8), B, 0, stream>>>(bufH, bufG, src, dst, dinv);
  k_post<H2P, H2R><<<cdiv((long long)N_NODES * H2P, B), B, 0, stream>>>(bufG, bufH, dinv, b2);

  // ---- classifier + log-softmax ----
  k_pack_a<<<cdiv((long long)MT * 4 * 32, B), B, 0, stream>>>(bufG, bufA, 4);
  k_gemm<3, 4><<<cdiv(MT, 8), B, 0, stream>>>(bufA, wf3, bufH);
  k_logsoftmax<<<cdiv(N_NODES, 8), B, 0, stream>>>(bufH, b3, out);
}
